// GraphConvModule_34677565948515
// MI455X (gfx1250) — matrix operand order repaired and checked by reading, not compiled, not run on hardware
//
#include <hip/hip_runtime.h>
#include <hip/hip_bf16.h>
#include <stdint.h>

#define N_NODES 50000
#define N_EDGES 1600000
#define IN_F 128
#define HID 256
#define ROWS_PAD 50176   // 196 blocks * 256 rows: padded row count for bf16 operand buffers

// keep prob = 0.4
#define KEEP_THRESH 1717986918u   // floor(0.4 * 2^32)
#define KEEP_SCALE  2.5f

typedef __attribute__((ext_vector_type(16))) __bf16 v16bf;
typedef __attribute__((ext_vector_type(8)))  float  v8f;

union BfVec {
  v16bf v;
  unsigned short u[16];
  uint4 q[2];
};

__device__ __forceinline__ unsigned short f32_to_bf16(float f) {
  unsigned u = __float_as_uint(f);
  u += 0x7FFFu + ((u >> 16) & 1u);        // round to nearest even
  return (unsigned short)(u >> 16);
}
__device__ __forceinline__ float bf16_to_f32(unsigned short s) {
  return __uint_as_float(((unsigned)s) << 16);
}
__device__ __forceinline__ unsigned hash32(unsigned x, unsigned seed) {
  x ^= seed;
  x *= 0x7feb352du; x ^= x >> 15;
  x *= 0x846ca68bu; x ^= x >> 16;
  return x;
}

// ---------------- dropout on input x -> h0 (f32 + bf16 hi/lo split) ----------------
__global__ void gc_dropout_kernel(const float* __restrict__ in, float* __restrict__ outf,
                                  unsigned short* __restrict__ outhi,
                                  unsigned short* __restrict__ outlo, int n, unsigned seed) {
  int i = blockIdx.x * blockDim.x + threadIdx.x;
  if (i < n) {
    float v = in[i];
    v = (hash32((unsigned)i, seed) < KEEP_THRESH) ? v * KEEP_SCALE : 0.0f;
    outf[i] = v;
    unsigned short hi = f32_to_bf16(v);
    outhi[i] = hi;
    outlo[i] = f32_to_bf16(v - bf16_to_f32(hi));
  }
}

// ---------------- f32 -> bf16 hi/lo split (for agg buffers after scatter) ----------------
__global__ void gc_split_kernel(const float* __restrict__ in,
                                unsigned short* __restrict__ outhi,
                                unsigned short* __restrict__ outlo, int n) {
  int i = blockIdx.x * blockDim.x + threadIdx.x;
  if (i < n) {
    float v = in[i];
    unsigned short hi = f32_to_bf16(v);
    outhi[i] = hi;
    outlo[i] = f32_to_bf16(v - bf16_to_f32(hi));
  }
}

// ---------------- pre-transpose weights to bf16 hi/lo, [N][Kcat] ----------------
__global__ void gc_prep_weights_kernel(const float* __restrict__ w_rel0,
                                       const float* __restrict__ w_root0,
                                       const float* __restrict__ w_rel1,
                                       const float* __restrict__ w_root1,
                                       unsigned short* __restrict__ wt0hi,
                                       unsigned short* __restrict__ wt0lo,
                                       unsigned short* __restrict__ wt1hi,
                                       unsigned short* __restrict__ wt1lo) {
  int i = blockIdx.x * blockDim.x + threadIdx.x;
  const int L0 = 256 * HID;   // Kcat0=256
  const int L1 = 512 * HID;   // Kcat1=512
  if (i < L0) {
    int k = i / HID, n = i % HID;
    float w = (k < IN_F) ? w_rel0[k * HID + n] : w_root0[(k - IN_F) * HID + n];
    unsigned short hi = f32_to_bf16(w);
    wt0hi[n * 256 + k] = hi;
    wt0lo[n * 256 + k] = f32_to_bf16(w - bf16_to_f32(hi));
  } else if (i < L0 + L1) {
    int j = i - L0;
    int k = j / HID, n = j % HID;
    float w = (k < HID) ? w_rel1[k * HID + n] : w_root1[(k - HID) * HID + n];
    unsigned short hi = f32_to_bf16(w);
    wt1hi[n * 512 + k] = hi;
    wt1lo[n * 512 + k] = f32_to_bf16(w - bf16_to_f32(hi));
  }
}

// ---------------- edge scatter-add: one wave per edge, hardware f32 atomics ----------------
template <int F>
__global__ void gc_scatter_kernel(const float* __restrict__ h,
                                  const long long* __restrict__ src,
                                  const long long* __restrict__ dst,
                                  float* __restrict__ agg) {
  int lane = threadIdx.x & 31;
  int e = blockIdx.x * (blockDim.x >> 5) + (threadIdx.x >> 5);
  if (e >= N_EDGES) return;
  long long s = src[e], d = dst[e];
  const float* hs = h + s * (long long)F;
  float* ad = agg + d * (long long)F;
#pragma unroll
  for (int c = lane * 4; c < F; c += 128) {
    float4 v = *(const float4*)(hs + c);
    unsafeAtomicAdd(ad + c + 0, v.x);
    unsafeAtomicAdd(ad + c + 1, v.y);
    unsafeAtomicAdd(ad + c + 2, v.z);
    unsafeAtomicAdd(ad + c + 3, v.w);
  }
}

// ---------------- fused GEMM: out = act([agg | h] @ [W_rel ; W_root] + b) ----------------
// A operands pre-split to bf16 hi/lo (A0 = agg part, A1 = h part, each [ROWS_PAD][KHALF]).
// B slab (64 N-columns of WT, all K, hi+lo) staged into LDS once per block via
// global_load_async_to_lds_b128; padded row stride -> conflict-free ds_load_b128.
// Block: 8 waves = 256(M) x 64(N); wave = 32(M: 2 tiles sharing B) x 64(N), 8 accumulators.
// Split-bf16 3-product scheme per tile; K-loop kept rolled to stay under 256 VGPRs.
template <int KCAT, bool DO_DROPOUT>
__global__ __launch_bounds__(256) void gc_gemm_kernel(
    const unsigned short* __restrict__ A0hi, const unsigned short* __restrict__ A0lo,
    const unsigned short* __restrict__ A1hi, const unsigned short* __restrict__ A1lo,
    const unsigned short* __restrict__ WThi, const unsigned short* __restrict__ WTlo,
    const float* __restrict__ bias, float* __restrict__ outf,
    unsigned short* __restrict__ outhi, unsigned short* __restrict__ outlo,
    int M, unsigned seed) {
  constexpr int KHALF = KCAT / 2;
  constexpr int LROW = KCAT + 8;      // padded LDS row stride: +4 banks/row -> conflict-free b128
  extern __shared__ unsigned short smem[];
  unsigned short* lbhi = smem;
  unsigned short* lblo = smem + 64 * LROW;

  const int tid = threadIdx.x;
  const int nbase = blockIdx.y * 64;

  // ---- stage B slab into LDS with CDNA5 async global->LDS copies ----
  {
    constexpr int CPR = KCAT / 8;     // 16B chunks per row
    constexpr int TOT = 64 * CPR;     // multiple of 256 -> uniform trip count, EXEC all-ones
    for (int c = tid; c < TOT; c += 256) {
      int row = c / CPR, j = c - row * CPR;
      unsigned lds_hi = (unsigned)(size_t)(lbhi + row * LROW + j * 8);
      unsigned lds_lo = (unsigned)(size_t)(lblo + row * LROW + j * 8);
      unsigned long long g_hi =
          (unsigned long long)(WThi + (size_t)(nbase + row) * KCAT + j * 8);
      unsigned long long g_lo =
          (unsigned long long)(WTlo + (size_t)(nbase + row) * KCAT + j * 8);
      asm volatile("global_load_async_to_lds_b128 %0, %1, off" ::"v"(lds_hi), "v"(g_hi)
                   : "memory");
      asm volatile("global_load_async_to_lds_b128 %0, %1, off" ::"v"(lds_lo), "v"(g_lo)
                   : "memory");
    }
    asm volatile("s_wait_asynccnt 0x0" ::: "memory");
    __syncthreads();
  }

  const int wave = tid >> 5;
  const int lane = tid & 31;
  const int lmod = lane & 15;
  const int lhi = lane >> 4;          // 0: lanes 0-15, 1: lanes 16-31
  const int c0 = lhi * 8;             // A layout: lanes 0-15 K{0..7,16..23}, 16-31 K{8..15,24..31}
  const long long mbase = (long long)(blockIdx.x * 8 + wave) * 32;
  const long long rowA0 = mbase + lmod;        // M tile 0
  const long long rowA1 = mbase + 16 + lmod;   // M tile 1 (padded buffers: loads always in-bounds)

  v8f acc[2][4] = {};

#pragma unroll 1
  for (int kb = 0; kb < KCAT; kb += 32) {
    const unsigned short* base_hi;
    const unsigned short* base_lo;
    long long koff;
    if (kb < KHALF) {
      base_hi = A0hi; base_lo = A0lo; koff = kb;
    } else {
      base_hi = A1hi; base_lo = A1lo; koff = kb - KHALF;
    }
    BfVec ahi0, alo0, ahi1, alo1;
    {
      const unsigned short* ph0 = base_hi + rowA0 * KHALF + koff;
      const unsigned short* pl0 = base_lo + rowA0 * KHALF + koff;
      const unsigned short* ph1 = base_hi + rowA1 * KHALF + koff;
      const unsigned short* pl1 = base_lo + rowA1 * KHALF + koff;
      ahi0.q[0] = *(const uint4*)(ph0 + c0);
      ahi0.q[1] = *(const uint4*)(ph0 + c0 + 16);
      alo0.q[0] = *(const uint4*)(pl0 + c0);
      alo0.q[1] = *(const uint4*)(pl0 + c0 + 16);
      ahi1.q[0] = *(const uint4*)(ph1 + c0);
      ahi1.q[1] = *(const uint4*)(ph1 + c0 + 16);
      alo1.q[0] = *(const uint4*)(pl1 + c0);
      alo1.q[1] = *(const uint4*)(pl1 + c0 + 16);
    }

    const int kB = kb + lhi * 8;      // B layout: lanes 0-15 K{0..7,16..23}, 16-31 K{8..15,24..31}
#pragma unroll
    for (int n = 0; n < 4; ++n) {
      const unsigned short* bh = lbhi + (n * 16 + lmod) * LROW + kB;
      const unsigned short* bl = lblo + (n * 16 + lmod) * LROW + kB;
      BfVec bhi, blo;
      bhi.q[0] = *(const uint4*)(bh);
      bhi.q[1] = *(const uint4*)(bh + 16);
      blo.q[0] = *(const uint4*)(bl);
      blo.q[1] = *(const uint4*)(bl + 16);
      acc[0][n] = __builtin_amdgcn_wmma_f32_16x16x32_bf16(false, ahi0.v, false, bhi.v,
                                                          (short)0, acc[0][n], false, false);
      acc[0][n] = __builtin_amdgcn_wmma_f32_16x16x32_bf16(false, alo0.v, false, bhi.v,
                                                          (short)0, acc[0][n], false, false);
      acc[0][n] = __builtin_amdgcn_wmma_f32_16x16x32_bf16(false, ahi0.v, false, blo.v,
                                                          (short)0, acc[0][n], false, false);
      acc[1][n] = __builtin_amdgcn_wmma_f32_16x16x32_bf16(false, ahi1.v, false, bhi.v,
                                                          (short)0, acc[1][n], false, false);
      acc[1][n] = __builtin_amdgcn_wmma_f32_16x16x32_bf16(false, alo1.v, false, bhi.v,
                                                          (short)0, acc[1][n], false, false);
      acc[1][n] = __builtin_amdgcn_wmma_f32_16x16x32_bf16(false, ahi1.v, false, blo.v,
                                                          (short)0, acc[1][n], false, false);
    }
  }

  // epilogue: C/D layout: lanes 0-15 -> M=j, lanes 16-31 -> M=j+8; N = nbase + n*16 + lmod
#pragma unroll
  for (int t = 0; t < 2; ++t) {
    long long tb = mbase + t * 16;
    if (tb >= M) continue;            // wave-uniform; tile base < M implies all 16 rows valid
#pragma unroll
    for (int n = 0; n < 4; ++n) {
      int col = nbase + n * 16 + lmod;
      float b = bias[col];
#pragma unroll
      for (int j = 0; j < 8; ++j) {
        long long row = tb + j + lhi * 8;
        float v = acc[t][n][j] + b;
        v = v > 0.0f ? v : 0.0f;
        if (DO_DROPOUT) {
          unsigned idx = (unsigned)(row * HID + col);
          v = (hash32(idx, seed) < KEEP_THRESH) ? v * KEEP_SCALE : 0.0f;
        }
        outf[row * HID + col] = v;
        if (DO_DROPOUT) {             // also emit next layer's A operand in bf16 hi/lo
          unsigned short hv = f32_to_bf16(v);
          outhi[row * HID + col] = hv;
          outlo[row * HID + col] = f32_to_bf16(v - bf16_to_f32(hv));
        }
      }
    }
  }
}

extern "C" void kernel_launch(void* const* d_in, const int* in_sizes, int n_in,
                              void* d_out, int out_size, void* d_ws, size_t ws_size,
                              hipStream_t stream) {
  (void)in_sizes; (void)n_in; (void)out_size; (void)ws_size;
  const float*     x       = (const float*)d_in[0];
  const long long* ei      = (const long long*)d_in[1];   // int64 [2][E]
  const float*     w_rel0  = (const float*)d_in[2];
  const float*     b_rel0  = (const float*)d_in[3];
  const float*     w_root0 = (const float*)d_in[4];
  const float*     w_rel1  = (const float*)d_in[5];
  const float*     b_rel1  = (const float*)d_in[6];
  const float*     w_root1 = (const float*)d_in[7];
  float* out = (float*)d_out;

  const long long* src = ei;
  const long long* dst = ei + N_EDGES;

  // ---- workspace layout (bf16 operand buffers padded to ROWS_PAD rows) ----
  char* ws = (char*)d_ws;
  size_t off = 0;
  auto alloc = [&](size_t bytes) -> char* {
    char* p = ws + off;
    off = (off + bytes + 255) & ~(size_t)255;
    return p;
  };
  unsigned short* wt0hi = (unsigned short*)alloc(256 * HID * 2);
  unsigned short* wt0lo = (unsigned short*)alloc(256 * HID * 2);
  unsigned short* wt1hi = (unsigned short*)alloc(512 * HID * 2);
  unsigned short* wt1lo = (unsigned short*)alloc(512 * HID * 2);
  float*          h0    = (float*)alloc((size_t)N_NODES * IN_F * 4);
  unsigned short* h0hi  = (unsigned short*)alloc((size_t)ROWS_PAD * IN_F * 2);
  unsigned short* h0lo  = (unsigned short*)alloc((size_t)ROWS_PAD * IN_F * 2);
  float*          agg0  = (float*)alloc((size_t)N_NODES * IN_F * 4);
  unsigned short* a0hi  = (unsigned short*)alloc((size_t)ROWS_PAD * IN_F * 2);
  unsigned short* a0lo  = (unsigned short*)alloc((size_t)ROWS_PAD * IN_F * 2);
  float*          h1    = (float*)alloc((size_t)N_NODES * HID * 4);
  unsigned short* h1hi  = (unsigned short*)alloc((size_t)ROWS_PAD * HID * 2);
  unsigned short* h1lo  = (unsigned short*)alloc((size_t)ROWS_PAD * HID * 2);
  float*          agg1  = (float*)alloc((size_t)N_NODES * HID * 4);
  unsigned short* a1hi  = (unsigned short*)alloc((size_t)ROWS_PAD * HID * 2);
  unsigned short* a1lo  = (unsigned short*)alloc((size_t)ROWS_PAD * HID * 2);

  // ---- weight prep (transpose + bf16 hi/lo split) ----
  {
    int total = 256 * HID + 512 * HID;
    gc_prep_weights_kernel<<<(total + 255) / 256, 256, 0, stream>>>(
        w_rel0, w_root0, w_rel1, w_root1, wt0hi, wt0lo, wt1hi, wt1lo);
  }

  // ---- layer 0 ----
  {
    int n = N_NODES * IN_F;
    gc_dropout_kernel<<<(n + 255) / 256, 256, 0, stream>>>(x, h0, h0hi, h0lo, n,
                                                           0x9E3779B9u);
  }
  hipMemsetAsync(agg0, 0, (size_t)N_NODES * IN_F * 4, stream);
  hipMemsetAsync(agg1, 0, (size_t)N_NODES * HID * 4, stream);
  {
    int blocks = N_EDGES / 8;   // 8 edges (waves) per 256-thread block
    gc_scatter_kernel<IN_F><<<blocks, 256, 0, stream>>>(h0, src, dst, agg0);
  }
  {
    int n = N_NODES * IN_F;
    gc_split_kernel<<<(n + 255) / 256, 256, 0, stream>>>(agg0, a0hi, a0lo, n);
  }
  {
    dim3 grid((N_NODES + 255) / 256, HID / 64);                  // (196, 4)
    size_t shmem = 2 * 64 * (256 + 8) * sizeof(unsigned short);  // 67,584 B
    gc_gemm_kernel<256, true><<<grid, 256, shmem, stream>>>(
        a0hi, a0lo, h0hi, h0lo, wt0hi, wt0lo, b_rel0, h1, h1hi, h1lo, N_NODES,
        0x85EBCA6Bu);
  }

  // ---- layer 1 ----
  {
    int blocks = N_EDGES / 8;
    gc_scatter_kernel<HID><<<blocks, 256, 0, stream>>>(h1, src, dst, agg1);
  }
  {
    int n = N_NODES * HID;
    gc_split_kernel<<<(n + 255) / 256, 256, 0, stream>>>(agg1, a1hi, a1lo, n);
  }
  {
    dim3 grid((N_NODES + 255) / 256, HID / 64);
    size_t shmem = 2 * 64 * (512 + 8) * sizeof(unsigned short);  // 133,120 B
    gc_gemm_kernel<512, false><<<grid, 256, shmem, stream>>>(
        a1hi, a1lo, h1hi, h1lo, wt1hi, wt1lo, b_rel1, out, (unsigned short*)0,
        (unsigned short*)0, N_NODES, 0u);
  }
}
